// CrossAttN_v8_60464549593375
// MI455X (gfx1250) — matrix operand order repaired and checked by reading, not compiled, not run on hardware
//
#include <hip/hip_runtime.h>

#define B_ 4
#define C_ 256
#define N_ 4096   // H*W = 64*64

typedef __bf16 bf16;
typedef __attribute__((ext_vector_type(16))) __bf16 v16bf;
typedef __attribute__((ext_vector_type(8)))  __bf16 v8bf;
typedef __attribute__((ext_vector_type(8)))  float  v8f;
typedef unsigned int u32x4 __attribute__((ext_vector_type(4)));
typedef int          i32x4 __attribute__((ext_vector_type(4)));
typedef int          i32x8 __attribute__((ext_vector_type(8)));

static __device__ __forceinline__ v8f wmma_bf16(v16bf a, v16bf b, v8f c) {
    // D = A(16x32 bf16) * B(32x16 bf16) + C(16x16 f32)
    return __builtin_amdgcn_wmma_f32_16x16x32_bf16(false, a, false, b, (short)0, c, false, false);
}

static __device__ __forceinline__ v16bf cat8(v8bf lo, v8bf hi) {
    return __builtin_shufflevector(lo, hi, 0,1,2,3,4,5,6,7,8,9,10,11,12,13,14,15);
}

// Exchange a v8bf (4 dwords) with the other 16-lane half of the wave.
static __device__ __forceinline__ v8bf shfl16_v8bf(v8bf x) {
    union { v8bf v; int i[4]; } u; u.v = x;
    #pragma unroll
    for (int k = 0; k < 4; ++k) u.i[k] = __shfl_xor(u.i[k], 16, 32);
    return u.v;
}

// Build a 32x16 B-operand (K=m) from two exp(S^T) 16x16 tile packs.
static __device__ __forceinline__ v16bf combineP(v8bf lo_t, v8bf hi_t, int half) {
    v8bf send = half ? lo_t : hi_t;
    v8bf recv = shfl16_v8bf(send);
    v8bf e0 = half ? recv : lo_t;   // low half: own K0..7 | high half: K16..23 from partner
    v8bf e1 = half ? hi_t : recv;   // low half: K8..15 from partner | high half: own K24..31
    return cat8(e0, e1);
}

// ---- Tensor Data Mover: async 2D tile load (global -> LDS), bf16 elements,
// row padding in LDS so row pitch = tile_d0 + 8 elements (16B per row).
// pad_interval code: (1<<code) qwords between pads; pad_amount code: (code+1) dwords.
static __device__ __forceinline__ void tdm_load_2d(unsigned int lds_addr, const void* gptr,
                                                   unsigned int tensor_d0, unsigned int tensor_d1,
                                                   unsigned int tile_d0, unsigned int tile_d1,
                                                   unsigned int stride0,
                                                   unsigned int pad_interval, unsigned int pad_amount) {
    unsigned long long ga = (unsigned long long)(uintptr_t)gptr;
    u32x4 g0;
    g0[0] = 1u;                                                   // count=1 (valid), user mode
    g0[1] = lds_addr;                                             // LDS byte address
    g0[2] = (unsigned int)(ga & 0xffffffffu);                     // global_addr[31:0]
    g0[3] = (unsigned int)((ga >> 32) & 0x01ffffffu) | (2u << 30);// global_addr[56:32] | type=2
    i32x8 g1;
    g1[0] = (int)((1u << 16) | (1u << 20) |                       // data_size=2B, pad_enable=1
                  (pad_interval << 22) | (pad_amount << 25));
    g1[1] = (int)((tensor_d0 & 0xffffu) << 16);                   // tensor_dim0[15:0] @ bit48
    g1[2] = (int)((tensor_d0 >> 16) | ((tensor_d1 & 0xffffu) << 16));
    g1[3] = (int)((tensor_d1 >> 16) | (tile_d0 << 16));           // tile_dim0 @ bit112
    g1[4] = (int)tile_d1;                                         // tile_dim1 @ bit128, tile_dim2=0
    g1[5] = (int)stride0;                                         // tensor_dim0_stride[31:0]
    g1[6] = 0;
    g1[7] = 0;
    i32x4 z4 = {0, 0, 0, 0};
#if __clang_major__ >= 23
    i32x8 z8 = {0, 0, 0, 0, 0, 0, 0, 0};
    __builtin_amdgcn_tensor_load_to_lds(g0, g1, z4, z4, z8, 0);
#else
    __builtin_amdgcn_tensor_load_to_lds(g0, g1, z4, z4, 0);
#endif
}

// ---------------- Kernel 0a: [B][C][N] f32 -> [B][N][C] bf16 (transpose + convert)
__global__ void k_transpose_cvt(const float* __restrict__ in, bf16* __restrict__ out) {
    __shared__ float tile[32][33];
    int bid = blockIdx.x;
    int nt = bid % (N_ / 32); bid /= (N_ / 32);
    int ct = bid % (C_ / 32);
    int b  = bid / (C_ / 32);
    int x  = threadIdx.x & 31;
    int y0 = threadIdx.x >> 5;   // 0..7
    const float* src = in + ((size_t)b * C_ + ct * 32) * N_ + nt * 32;
    #pragma unroll
    for (int i = 0; i < 4; ++i) {
        int y = y0 + i * 8;                       // c-local
        tile[y][x] = src[(size_t)y * N_ + x];
    }
    __syncthreads();
    bf16* dst = out + ((size_t)b * N_ + nt * 32) * C_ + ct * 32;
    #pragma unroll
    for (int i = 0; i < 4; ++i) {
        int y = y0 + i * 8;                       // n-local
        dst[(size_t)y * C_ + x] = (bf16)tile[x][y];
    }
}

// ---------------- Kernel 0b: convert 4 weight matrices f32 -> bf16
__global__ void k_cvt_weights(const float* __restrict__ wf, const float* __restrict__ wg,
                              const float* __restrict__ wh, const float* __restrict__ wo,
                              bf16* __restrict__ of, bf16* __restrict__ og,
                              bf16* __restrict__ oh, bf16* __restrict__ oo) {
    int i = blockIdx.x * blockDim.x + threadIdx.x;
    if (i < C_ * C_) {
        of[i] = (bf16)wf[i];
        og[i] = (bf16)wg[i];
        oh[i] = (bf16)wh[i];
        oo[i] = (bf16)wo[i];
    }
}

// ---------------- Kernel 1: projection  Out[d,n] = sum_c W[d,c]*Act[n,c] + bias[d]
// act: [B][N][C] bf16 (n-major).  OUT_CMAJOR=0 -> out bf16 [B][N][C]; =1 -> out bf16 [B][C][N]
template<int OUT_CMAJOR>
__global__ void __launch_bounds__(128) k_proj(const bf16* __restrict__ act,
                                              const bf16* __restrict__ wt,
                                              const float* __restrict__ bias,
                                              bf16* __restrict__ out) {
    int wave  = blockIdx.x * (blockDim.x >> 5) + (threadIdx.x >> 5);
    int lane  = threadIdx.x & 31;
    int half  = lane >> 4;
    int lrow  = lane & 15;
    int ntile = wave % (N_ / 16);
    int b     = wave / (N_ / 16);
    int n0    = ntile * 16;

    v16bf bq[8];
    const bf16* arow = act + ((size_t)b * N_ + n0 + lrow) * C_ + half * 16;
    #pragma unroll
    for (int kc = 0; kc < 8; ++kc) bq[kc] = *(const v16bf*)(arow + kc * 32);

    for (int dt = 0; dt < 16; ++dt) {
        int dbase = dt * 16 + half * 8;
        v8f acc;
        #pragma unroll
        for (int v = 0; v < 8; ++v) acc[v] = bias[dbase + v];
        const bf16* wrow = wt + (size_t)(dt * 16 + lrow) * C_;
        #pragma unroll
        for (int kc = 0; kc < 8; ++kc) {
            v8bf a0 = *(const v8bf*)(wrow + kc * 32 + half * 16);
            v8bf a1 = *(const v8bf*)(wrow + kc * 32 + half * 16 + 8);
            acc = wmma_bf16(cat8(a0, a1), bq[kc], acc);
        }
        if (OUT_CMAJOR) {
            int m = n0 + lrow;
            #pragma unroll
            for (int v = 0; v < 8; ++v)
                out[((size_t)b * C_ + dbase + v) * N_ + m] = (bf16)acc[v];
        } else {
            v8bf o;
            #pragma unroll
            for (int v = 0; v < 8; ++v) o[v] = (bf16)acc[v];
            *(v8bf*)(out + ((size_t)b * N_ + n0 + lrow) * C_ + dbase) = o;
        }
    }
}

// ---------------- Kernel 2: fused flash attention with TDM double-buffered staging
#define MSTEP 64
__global__ void __launch_bounds__(128, 1) k_attn(const bf16* __restrict__ Fqt, // [B][N][C]
                                                 const bf16* __restrict__ Gt,  // [B][N][C] m-major
                                                 const bf16* __restrict__ Hvc, // [B][C][N]
                                                 bf16* __restrict__ rest) {    // [B][N][C]
    __shared__ bf16 Gs[2][MSTEP][C_ + 8];     // TDM-padded: row pitch 528B
    __shared__ bf16 Hs[2][C_][MSTEP + 8];     // TDM-padded: row pitch 144B
    int wid  = threadIdx.x >> 5;
    int lane = threadIdx.x & 31;
    int half = lane >> 4;
    int lrow = lane & 15;
    int blocksPerB = (N_ / 16) / 4;        // 64
    int b  = blockIdx.x / blocksPerB;
    int qc = blockIdx.x % blocksPerB;
    int n0 = qc * 64 + wid * 16;

    const bf16* gbase = Gt  + (size_t)b * N_ * C_;
    const bf16* hbase = Hvc + (size_t)b * C_ * N_;

    // Resident query B operands Fq[c, n0..n0+15]
    v16bf bq[8];
    const bf16* qrow = Fqt + ((size_t)b * N_ + n0 + lrow) * C_ + half * 16;
    #pragma unroll
    for (int kc = 0; kc < 8; ++kc) bq[kc] = *(const v16bf*)(qrow + kc * 32);

    v8f acc[16];
    #pragma unroll
    for (int t = 0; t < 16; ++t) acc[t] = v8f{};
    float m_r = -3.0e38f, l_r = 0.0f;

    // Prologue: issue chunk 0 into buffer 0 (one wave drives the TDM)
    if (wid == 0) {
        tdm_load_2d((unsigned int)(uintptr_t)&Gs[0][0][0], gbase,
                    C_, N_, C_, MSTEP, C_, /*interval 128dw*/6, /*pad 4dw*/3);
        tdm_load_2d((unsigned int)(uintptr_t)&Hs[0][0][0], hbase,
                    N_, C_, MSTEP, C_, N_, /*interval 32dw*/4, /*pad 4dw*/3);
    }

    for (int m0 = 0; m0 < N_; m0 += MSTEP) {
        int buf = (m0 / MSTEP) & 1;
        if (wid == 0) {
            if (m0 + MSTEP < N_) {
                // Issue next chunk into the other buffer (safe: all waves passed the
                // trailing barrier of the previous iteration before we get here).
                tdm_load_2d((unsigned int)(uintptr_t)&Gs[buf ^ 1][0][0],
                            gbase + (size_t)(m0 + MSTEP) * C_,
                            C_, N_ - (m0 + MSTEP), C_, MSTEP, C_, 6, 3);
                tdm_load_2d((unsigned int)(uintptr_t)&Hs[buf ^ 1][0][0],
                            hbase + (m0 + MSTEP),
                            N_ - (m0 + MSTEP), C_, MSTEP, C_, N_, 4, 3);
                __builtin_amdgcn_s_wait_tensorcnt(2);  // in-order: current chunk landed
            } else {
                __builtin_amdgcn_s_wait_tensorcnt(0);
            }
        }
        __syncthreads();   // current chunk visible to all waves

        // 4 S^T tiles: S^T[m, n] over m in [m0, m0+64)
        v8f st[4];
        #pragma unroll
        for (int t = 0; t < 4; ++t) {
            v8f s = v8f{};
            const bf16* grow = &Gs[buf][t * 16 + lrow][0];
            #pragma unroll
            for (int kc = 0; kc < 8; ++kc) {
                v8bf a0 = *(const v8bf*)(grow + kc * 32 + half * 16);
                v8bf a1 = *(const v8bf*)(grow + kc * 32 + half * 16 + 8);
                s = wmma_bf16(cat8(a0, a1), bq[kc], s);
            }
            st[t] = s;
        }

        // Online softmax over this 64-key block
        float vmax = st[0][0];
        #pragma unroll
        for (int t = 0; t < 4; ++t)
            #pragma unroll
            for (int v = 0; v < 8; ++v) vmax = fmaxf(vmax, st[t][v]);
        vmax = fmaxf(vmax, __shfl_xor(vmax, 16, 32));
        float newm  = fmaxf(m_r, vmax);
        float alpha = __expf(m_r - newm);
        m_r = newm;
        #pragma unroll
        for (int t = 0; t < 16; ++t)
            #pragma unroll
            for (int v = 0; v < 8; ++v) acc[t][v] *= alpha;
        float psum = 0.0f;
        v8bf pt[4];
        #pragma unroll
        for (int t = 0; t < 4; ++t)
            #pragma unroll
            for (int v = 0; v < 8; ++v) {
                float p = __expf(st[t][v] - m_r);
                psum += p;
                pt[t][v] = (bf16)p;
            }
        l_r = l_r * alpha + psum + __shfl_xor(psum, 16, 32);

        v16bf B0 = combineP(pt[0], pt[1], half);   // keys m0..m0+31
        v16bf B1 = combineP(pt[2], pt[3], half);   // keys m0+32..m0+63

        // acc[ct] (16c x 16n) += Hv(16c x 32m) * P(32m x 16n)
        #pragma unroll
        for (int ct = 0; ct < 16; ++ct) {
            const bf16* hrow = &Hs[buf][ct * 16 + lrow][0];
            v8bf a0 = *(const v8bf*)(hrow + half * 16);
            v8bf a1 = *(const v8bf*)(hrow + half * 16 + 8);
            acc[ct] = wmma_bf16(cat8(a0, a1), B0, acc[ct]);
            v8bf a2 = *(const v8bf*)(hrow + 32 + half * 16);
            v8bf a3 = *(const v8bf*)(hrow + 40 + half * 16);
            acc[ct] = wmma_bf16(cat8(a2, a3), B1, acc[ct]);
        }
        __syncthreads();   // everyone done reading buf before it is re-filled
    }

    float inv = 1.0f / l_r;
    bf16* orow = rest + ((size_t)b * N_ + n0 + lrow) * C_;
    #pragma unroll
    for (int ct = 0; ct < 16; ++ct) {
        v8bf o;
        #pragma unroll
        for (int v = 0; v < 8; ++v) o[v] = (bf16)(acc[ct][v] * inv);
        *(v8bf*)(orow + ct * 16 + half * 8) = o;
    }
}

// ---------------- Kernel 3: output projection, f32 c-major out
__global__ void __launch_bounds__(128) k_out(const bf16* __restrict__ rest, // [B][N][C]
                                             const bf16* __restrict__ wo,
                                             const float* __restrict__ bo,
                                             float* __restrict__ out) {     // [B][C][N] f32
    int wave  = blockIdx.x * (blockDim.x >> 5) + (threadIdx.x >> 5);
    int lane  = threadIdx.x & 31;
    int half  = lane >> 4;
    int lrow  = lane & 15;
    int ntile = wave % (N_ / 16);
    int b     = wave / (N_ / 16);
    int n0    = ntile * 16;

    v16bf bq[8];
    const bf16* arow = rest + ((size_t)b * N_ + n0 + lrow) * C_ + half * 16;
    #pragma unroll
    for (int kc = 0; kc < 8; ++kc) bq[kc] = *(const v16bf*)(arow + kc * 32);

    for (int dt = 0; dt < 16; ++dt) {
        int dbase = dt * 16 + half * 8;
        v8f acc;
        #pragma unroll
        for (int v = 0; v < 8; ++v) acc[v] = bo[dbase + v];
        const bf16* wrow = wo + (size_t)(dt * 16 + lrow) * C_;
        #pragma unroll
        for (int kc = 0; kc < 8; ++kc) {
            v8bf a0 = *(const v8bf*)(wrow + kc * 32 + half * 16);
            v8bf a1 = *(const v8bf*)(wrow + kc * 32 + half * 16 + 8);
            acc = wmma_bf16(cat8(a0, a1), bq[kc], acc);
        }
        int n = n0 + lrow;
        #pragma unroll
        for (int v = 0; v < 8; ++v)
            out[((size_t)b * C_ + dbase + v) * N_ + n] = acc[v];
    }
}

extern "C" void kernel_launch(void* const* d_in, const int* in_sizes, int n_in,
                              void* d_out, int out_size, void* d_ws, size_t ws_size,
                              hipStream_t stream) {
    (void)in_sizes; (void)n_in; (void)out_size; (void)ws_size;
    const float* F_c = (const float*)d_in[0];
    const float* F_s = (const float*)d_in[1];
    const float* Wf  = (const float*)d_in[2];
    const float* bf_ = (const float*)d_in[3];
    const float* Wg  = (const float*)d_in[4];
    const float* bg  = (const float*)d_in[5];
    const float* Wh  = (const float*)d_in[6];
    const float* bh  = (const float*)d_in[7];
    const float* Wo  = (const float*)d_in[8];
    const float* bo  = (const float*)d_in[9];
    float* out = (float*)d_out;

    const size_t NB = (size_t)B_ * N_ * C_;
    bf16* p   = (bf16*)d_ws;
    bf16* Fct = p; p += NB;
    bf16* Fst = p; p += NB;
    bf16* Fqt = p; p += NB;
    bf16* Gt  = p; p += NB;
    bf16* Hvc = p; p += NB;
    bf16* res = p; p += NB;
    bf16* Wfb = p; p += (size_t)C_ * C_;
    bf16* Wgb = p; p += (size_t)C_ * C_;
    bf16* Whb = p; p += (size_t)C_ * C_;
    bf16* Wob = p; p += (size_t)C_ * C_;

    k_transpose_cvt<<<B_ * (C_ / 32) * (N_ / 32), 256, 0, stream>>>(F_c, Fct);
    k_transpose_cvt<<<B_ * (C_ / 32) * (N_ / 32), 256, 0, stream>>>(F_s, Fst);
    k_cvt_weights<<<(C_ * C_) / 256, 256, 0, stream>>>(Wf, Wg, Wh, Wo, Wfb, Wgb, Whb, Wob);

    k_proj<0><<<256, 128, 0, stream>>>(Fct, Wfb, bf_, Fqt);   // Fq, n-major
    k_proj<0><<<256, 128, 0, stream>>>(Fst, Wgb, bg, Gt);     // G,  m-major
    k_proj<1><<<256, 128, 0, stream>>>(Fst, Whb, bh, Hvc);    // Hv, c-major

    k_attn<<<256, 128, 0, stream>>>(Fqt, Gt, Hvc, res);
    k_out<<<256, 128, 0, stream>>>(res, Wob, bo, out);
}
